// DemodConv_61641370632560
// MI455X (gfx1250) — matrix operand order repaired and checked by reading, not compiled, not run on hardware
//
#include <hip/hip_runtime.h>
#include <hip/hip_bf16.h>

// ---------------------------------------------------------------------------
// DemodConv (StyleGAN2 modulated conv) for MI455X / gfx1250.
// b=8, cin=512, cout=512, k=3, h=w=32 (fixed by reference setup_inputs()).
//
// Compute-bound (AI ~ 900 FLOP/B) -> implicit-GEMM on the bf16 WMMA pipe
// (V_WMMA_F32_16X16X32_BF16, fp32 accumulate). Per-batch GEMM:
//   M = cout(512), N = h*w(1024), K = cin*9(4608)
// Modulation (s+1) and demodulation fused into the A-tile stage; im2col of x
// fused into the B-tile stage. Double-buffered LDS, one barrier per K-step.
// Both LDS tiles laid out so every WMMA fragment is two aligned 16B LDS
// loads (no v_mov_b16 repacking): A row-major [M][K], B transposed [N][K].
// ---------------------------------------------------------------------------

typedef __attribute__((ext_vector_type(16))) __bf16 v16bf;
typedef __attribute__((ext_vector_type(8)))  __bf16 v8bf;
typedef __attribute__((ext_vector_type(8)))  float  v8f;

#define B_    8
#define CIN   512
#define COUT  512
#define HW    1024        // 32*32 spatial
#define KTOT  4608        // cin * 9
#define BM    128
#define BN    128
#define BK    32
#define PAD   8           // row stride 40 bf16 = 80B: 16B-aligned, 20-bank step
#define EPS_  1e-8f

// --- Kernel 1: per-(b,o) demodulation scale -------------------------------
__global__ __launch_bounds__(256)
void demod_kernel(const float* __restrict__ w,   // (cout, cin*9)
                  const float* __restrict__ s,   // (b, cin)
                  float* __restrict__ demod)     // (b, cout)
{
    const int bo = blockIdx.x;           // b*COUT + o
    const int b  = bo >> 9;
    const int o  = bo & (COUT - 1);
    const float* wrow = w + (size_t)o * KTOT;
    const float* srow = s + (size_t)b * CIN;

    float acc = 0.0f;
    for (int idx = threadIdx.x; idx < KTOT; idx += 256) {
        const int   ci = idx / 9;
        const float v  = wrow[idx] * (srow[ci] + 1.0f);
        acc += v * v;
    }
    __shared__ float red[256];
    red[threadIdx.x] = acc;
    __syncthreads();
    #pragma unroll
    for (int off = 128; off > 0; off >>= 1) {
        if (threadIdx.x < off) red[threadIdx.x] += red[threadIdx.x + off];
        __syncthreads();
    }
    if (threadIdx.x == 0) demod[bo] = rsqrtf(red[0] + EPS_);
}

// --- Kernel 2: implicit-GEMM modulated conv on the WMMA pipe --------------
__global__ __launch_bounds__(256)
void modconv_kernel(const float* __restrict__ x,      // (b, cin, 32, 32)
                    const float* __restrict__ s,      // (b, cin)
                    const float* __restrict__ w,      // (cout, cin*9)
                    const float* __restrict__ demod,  // (b, cout)
                    float* __restrict__ out)          // (b, cout, 32, 32)
{
    const int b     = blockIdx.z;
    const int nBase = blockIdx.x * BN;   // spatial tile
    const int oBase = blockIdx.y * BM;   // output-channel tile

    // Double-buffered tiles; 80B row stride (aligned 16B, conflict-free).
    __shared__ __bf16 ldsA [2][BM][BK + PAD];  // weights, [M][K]
    __shared__ __bf16 ldsBt[2][BN][BK + PAD];  // im2col x, TRANSPOSED [N][K]

    const int tid  = threadIdx.x;
    const int lane = tid & 31;           // wave32
    const int wv   = tid >> 5;           // 8 waves
    const int wm   = wv & 3;             // 4 waves along M (32 rows each)
    const int wn   = wv >> 2;            // 2 waves along N (64 cols each)

    v8f acc[2][4] = {};                  // 2x4 WMMA tiles, fp32 accum

    // Cooperative-load mapping: one integer-div set per thread per stage.
    const int a_kk = tid & 31;           // A: fixed K column, 16 M rows
    const int a_m0 = (tid >> 5) * 16;
    const int b_r  = tid & 31;           // B: fixed K row, 16 N cols
    const int b_c0 = (tid >> 5) * 16;

    const float* xb = x + (size_t)b * CIN * HW;
    const float* sb = s + (size_t)b * CIN;
    const float* db = demod + (size_t)b * COUT;

    // Fragment gather coordinates (CDNA5 wave32 WMMA VGPR striping).
    const int fm  = lane & 15;           // A row / B column within tile
    const int fkh = lane >> 4;           // K-half select

    // ---- tile stagers (fused modulation / im2col, fp32 -> bf16) ----
    auto stageA = [&](int buf, int k0) {
        const int   kidx = k0 + a_kk;
        const int   ci   = kidx / 9;
        const float sm   = sb[ci] + 1.0f;
        const float* wp  = w + (size_t)(oBase + a_m0) * KTOT + kidx;
        #pragma unroll
        for (int mm = 0; mm < 16; ++mm) {
            const float v = wp[(size_t)mm * KTOT] * sm * db[oBase + a_m0 + mm];
            ldsA[buf][a_m0 + mm][a_kk] = (__bf16)v;
        }
    };
    auto stageB = [&](int buf, int k0) {
        const int kidx = k0 + b_r;
        const int ci   = kidx / 9;
        const int t    = kidx - ci * 9;
        const int dy   = t / 3;
        const int dx   = t - dy * 3;
        const float* xc = xb + (size_t)ci * HW;
        #pragma unroll
        for (int cc = 0; cc < 16; ++cc) {
            const int p  = nBase + b_c0 + cc;
            const int ho = p >> 5, wo = p & 31;
            const int hi = ho + dy - 1, wi = wo + dx - 1;
            float v = 0.0f;
            if ((unsigned)hi < 32u && (unsigned)wi < 32u)
                v = xc[hi * 32 + wi];
            ldsBt[buf][b_c0 + cc][b_r] = (__bf16)v;   // transposed store
        }
    };

    stageA(0, 0);
    stageB(0, 0);
    __syncthreads();

    int cur = 0;
    for (int k0 = 0; k0 < KTOT; k0 += BK) {
        // ---- gather fragments: two aligned 16B LDS loads each ----
        v16bf afrag[2], bfrag[4];
        #pragma unroll
        for (int tm = 0; tm < 2; ++tm) {
            const int mrow = wm * 32 + tm * 16 + fm;
            const v8bf alo = *(const v8bf*)&ldsA[cur][mrow][fkh * 8];      // K 0-7 / 8-15
            const v8bf ahi = *(const v8bf*)&ldsA[cur][mrow][16 + fkh * 8]; // K 16-23 / 24-31
            afrag[tm] = __builtin_shufflevector(alo, ahi,
                0,1,2,3,4,5,6,7,8,9,10,11,12,13,14,15);
        }
        #pragma unroll
        for (int tn = 0; tn < 4; ++tn) {
            const int ncol = wn * 64 + tn * 16 + fm;
            const v8bf blo = *(const v8bf*)&ldsBt[cur][ncol][fkh * 8];      // K 0-7 / 8-15
            const v8bf bhi = *(const v8bf*)&ldsBt[cur][ncol][16 + fkh * 8]; // K 16-23 / 24-31
            bfrag[tn] = __builtin_shufflevector(blo, bhi,
                0,1,2,3,4,5,6,7,8,9,10,11,12,13,14,15);
        }

        // ---- overlap: stage next tile into the other buffer ----
        const int knext = k0 + BK;
        if (knext < KTOT) {
            stageA(cur ^ 1, knext);
            stageB(cur ^ 1, knext);
        }

        // ---- 8 WMMAs per wave per K step ----
        #pragma unroll
        for (int tm = 0; tm < 2; ++tm)
            #pragma unroll
            for (int tn = 0; tn < 4; ++tn)
                acc[tm][tn] = __builtin_amdgcn_wmma_f32_16x16x32_bf16(
                    /*neg_a=*/false, afrag[tm],
                    /*neg_b=*/false, bfrag[tn],
                    /*c_mod=*/(short)0, acc[tm][tn],
                    /*reuse_a=*/false, /*reuse_b=*/false);

        __syncthreads();   // next-buffer writes visible; this-buffer reads done
        cur ^= 1;
    }

    // ---- epilogue: C/D layout (VGPR r -> M=r / M=8+r) -> out[b][o][p] ----
    float* ob = out + (size_t)b * COUT * HW;
    const int Mofs = (lane >> 4) * 8;
    const int Ncol = lane & 15;
    #pragma unroll
    for (int tm = 0; tm < 2; ++tm)
        #pragma unroll
        for (int tn = 0; tn < 4; ++tn) {
            const int oRow = oBase + wm * 32 + tm * 16 + Mofs;
            const int p    = nBase + wn * 64 + tn * 16 + Ncol;
            #pragma unroll
            for (int r = 0; r < 8; ++r)
                ob[(size_t)(oRow + r) * HW + p] = acc[tm][tn][r];
        }
}

extern "C" void kernel_launch(void* const* d_in, const int* in_sizes, int n_in,
                              void* d_out, int out_size, void* d_ws, size_t ws_size,
                              hipStream_t stream) {
    (void)in_sizes; (void)n_in; (void)out_size; (void)ws_size;
    const float* x = (const float*)d_in[0];   // (8, 512, 32, 32)
    const float* s = (const float*)d_in[1];   // (8, 512)
    const float* w = (const float*)d_in[2];   // (512, 512, 3, 3)
    float* out   = (float*)d_out;             // (8, 512, 32, 32)
    float* demod = (float*)d_ws;              // 8*512 floats = 16 KB scratch

    demod_kernel<<<B_ * COUT, 256, 0, stream>>>(w, s, demod);

    dim3 grid(HW / BN, COUT / BM, B_);        // 8 x 4 x 8 = 256 workgroups
    modconv_kernel<<<grid, 256, 0, stream>>>(x, s, w, demod, out);
}
